// MoeLayer_14379550507738
// MI455X (gfx1250) — matrix operand order repaired and checked by reading, not compiled, not run on hardware
//
#include <hip/hip_runtime.h>
#include <hip/hip_bf16.h>

typedef __attribute__((ext_vector_type(16))) __bf16 v16bf;
typedef __attribute__((ext_vector_type(8)))  __bf16 v8bf;
typedef __attribute__((ext_vector_type(8)))  float  v8f;

#define T_TOK 8192   // B*S tokens
#define DIMD  1024
#define NEXP  8
#define DIMF  4096
#define CAP   1024   // capacity = T/E

// ---------------------------------------------------------------- router ----
__global__ __launch_bounds__(256) void router_kernel(
    const float* __restrict__ x, const float* __restrict__ rw,
    float* __restrict__ gate, int* __restrict__ eidx)
{
  int tok  = (blockIdx.x * blockDim.x + threadIdx.x) >> 5;  // one wave per token
  int lane = threadIdx.x & 31;
  if (tok >= T_TOK) return;
  const float* xr = x + (long)tok * DIMD;
  float acc[NEXP];
#pragma unroll
  for (int e = 0; e < NEXP; ++e) acc[e] = 0.f;
  for (int d = lane; d < DIMD; d += 32) {
    float xv = xr[d];
    const float4* r4 = (const float4*)(rw + d * NEXP);
    float4 r0 = r4[0], r1 = r4[1];
    acc[0] += xv * r0.x; acc[1] += xv * r0.y; acc[2] += xv * r0.z; acc[3] += xv * r0.w;
    acc[4] += xv * r1.x; acc[5] += xv * r1.y; acc[6] += xv * r1.z; acc[7] += xv * r1.w;
  }
#pragma unroll
  for (int off = 16; off > 0; off >>= 1) {
#pragma unroll
    for (int e = 0; e < NEXP; ++e) acc[e] += __shfl_xor(acc[e], off, 32);
  }
  if (lane == 0) {
    float m = acc[0]; int mi = 0;
#pragma unroll
    for (int e = 1; e < NEXP; ++e) if (acc[e] > m) { m = acc[e]; mi = e; }
    float s = 0.f;
#pragma unroll
    for (int e = 0; e < NEXP; ++e) s += __expf(acc[e] - m);
    gate[tok] = 1.f / s;   // max softmax prob
    eidx[tok] = mi;
  }
}

// ------------------------------------------------- ordered dispatch scan ----
__global__ __launch_bounds__(256) void scan_kernel(
    const int* __restrict__ eidx, int* __restrict__ pos, int* __restrict__ slot_tok)
{
  __shared__ int cnt[256][NEXP];
  __shared__ int total[NEXP];
  const int tid = threadIdx.x;
  const int per = T_TOK / 256;       // 32 consecutive tokens per thread
  const int t0  = tid * per;

  int c[NEXP];
#pragma unroll
  for (int e = 0; e < NEXP; ++e) c[e] = 0;
  for (int i = 0; i < per; ++i) c[eidx[t0 + i]]++;
#pragma unroll
  for (int e = 0; e < NEXP; ++e) cnt[tid][e] = c[e];
  __syncthreads();

  if (tid == 0) {                    // serial exclusive scan (2048 adds)
    int run[NEXP];
    for (int e = 0; e < NEXP; ++e) run[e] = 0;
    for (int i = 0; i < 256; ++i)
      for (int e = 0; e < NEXP; ++e) { int v = cnt[i][e]; cnt[i][e] = run[e]; run[e] += v; }
    for (int e = 0; e < NEXP; ++e) total[e] = run[e];
  }
  __syncthreads();

  int base[NEXP];
#pragma unroll
  for (int e = 0; e < NEXP; ++e) base[e] = cnt[tid][e];
  for (int i = 0; i < per; ++i) {
    int t = t0 + i;
    int e = eidx[t];
    int p = base[e]++;
    pos[t] = p;
    if (p < CAP) slot_tok[e * CAP + p] = t;
  }
  // mark never-filled slots
  for (int s = tid; s < NEXP * CAP; s += 256) {
    int e = s / CAP, cc = s % CAP;
    if (cc >= total[e]) slot_tok[s] = -1;
  }
}

// ---------------------------------------------------------------- gather ----
__global__ __launch_bounds__(256) void gather_kernel(
    const float* __restrict__ x, const int* __restrict__ slot_tok, float* __restrict__ xe)
{
  int slot = blockIdx.x;
  int t = slot_tok[slot];
  float4* dst = (float4*)(xe + (long)slot * DIMD);
  if (t >= 0) {
    const float4* src = (const float4*)(x + (long)t * DIMD);
    dst[threadIdx.x] = src[threadIdx.x];
  } else {
    dst[threadIdx.x] = float4{0.f, 0.f, 0.f, 0.f};
  }
}

// ------------------------------------------- split-bf16 WMMA batched GEMM ---
// C[e] = (optional relu)(A[e](MxK) * B[e](KxN) + bias[e][n]), all row-major f32.
// Double-buffered LDS; one barrier per K-chunk; global prefetch under WMMAs.
#define BM 128
#define BN 128
#define KC 32
#define KPAD 40   // 80-byte row stride -> conflict-free b128 fragment loads

__global__ __launch_bounds__(256) void gemm_kernel(
    const float* __restrict__ Aall, const float* __restrict__ Ball,
    float* __restrict__ Call, const float* __restrict__ biasAll,
    int N, int K, long sA, long sB, long sC, long sBias, int do_relu)
{
  __shared__ __attribute__((aligned(16))) __bf16 Ah[2][BM][KPAD];
  __shared__ __attribute__((aligned(16))) __bf16 Al[2][BM][KPAD];
  __shared__ __attribute__((aligned(16))) __bf16 Bh[2][BN][KPAD];  // transposed [n][k]
  __shared__ __attribute__((aligned(16))) __bf16 Bl[2][BN][KPAD];

  const int e = blockIdx.z;
  const float* A  = Aall + (long)e * sA;
  const float* Bg = Ball + (long)e * sB;
  float*       C  = Call + (long)e * sC;
  const float* bias = biasAll + (long)e * sBias;

  const int bn0 = blockIdx.x * BN;
  const int bm0 = blockIdx.y * BM;

  const int tid  = threadIdx.x;
  const int lane = tid & 31;
  const int wid  = tid >> 5;
  const int wm   = wid & 3;     // 4 waves along M -> 32 rows each
  const int wn   = wid >> 2;    // 2 waves along N -> 64 cols each
  const int lm   = lane & 15;
  const int lh   = lane >> 4;

  // per-thread staging coordinates (4 float4 per matrix per chunk)
  const int arow[4] = { (0*256+tid) >> 3, (1*256+tid) >> 3, (2*256+tid) >> 3, (3*256+tid) >> 3 };
  const int ac4     = (tid & 7) * 4;
  const int bkr[4]  = { (0*256+tid) >> 5, (1*256+tid) >> 5, (2*256+tid) >> 5, (3*256+tid) >> 5 };
  const int bn4     = (tid & 31) * 4;

  float4 pa[4], pb[4];
  auto gload = [&](int kc) {
#pragma unroll
    for (int p = 0; p < 4; ++p) {
      pa[p] = *(const float4*)(A  + (long)(bm0 + arow[p]) * K + kc + ac4);
      pb[p] = *(const float4*)(Bg + (long)(kc + bkr[p]) * N + bn0 + bn4);
    }
  };
  auto sstore = [&](int buf) {
#pragma unroll
    for (int p = 0; p < 4; ++p) {
      float va[4] = { pa[p].x, pa[p].y, pa[p].z, pa[p].w };
      float vb[4] = { pb[p].x, pb[p].y, pb[p].z, pb[p].w };
#pragma unroll
      for (int i = 0; i < 4; ++i) {
        __bf16 h = (__bf16)va[i];
        Ah[buf][arow[p]][ac4 + i] = h;
        Al[buf][arow[p]][ac4 + i] = (__bf16)(va[i] - (float)h);
        __bf16 g = (__bf16)vb[i];
        Bh[buf][bn4 + i][bkr[p]] = g;
        Bl[buf][bn4 + i][bkr[p]] = (__bf16)(vb[i] - (float)g);
      }
    }
  };

  v8f acc[2][4];
#pragma unroll
  for (int i = 0; i < 2; ++i)
#pragma unroll
    for (int j = 0; j < 4; ++j)
#pragma unroll
      for (int r = 0; r < 8; ++r) acc[i][j][r] = 0.f;

  // prologue: fill buffer 0
  gload(0);
  sstore(0);
  __syncthreads();

  const int nk = K / KC;
  for (int k = 0; k < nk; ++k) {
    const int cur = k & 1;
    if (k + 1 < nk) gload((k + 1) * KC);   // prefetch next chunk (hidden under WMMAs)

    // ---- fragments (ISA 16-bit A 16x32 layout; B from transposed tile) ----
    v16bf afh[2], afl[2], bfh[4], bfl[4];
#pragma unroll
    for (int mt = 0; mt < 2; ++mt) {
      int row = wm * 32 + mt * 16 + lm;
      v8bf p0 = *(const v8bf*)&Ah[cur][row][8 * lh];
      v8bf p1 = *(const v8bf*)&Ah[cur][row][16 + 8 * lh];
      afh[mt] = __builtin_shufflevector(p0, p1, 0,1,2,3,4,5,6,7,8,9,10,11,12,13,14,15);
      v8bf q0 = *(const v8bf*)&Al[cur][row][8 * lh];
      v8bf q1 = *(const v8bf*)&Al[cur][row][16 + 8 * lh];
      afl[mt] = __builtin_shufflevector(q0, q1, 0,1,2,3,4,5,6,7,8,9,10,11,12,13,14,15);
    }
#pragma unroll
    for (int nt = 0; nt < 4; ++nt) {
      int col = wn * 64 + nt * 16 + lm;
      v8bf p0 = *(const v8bf*)&Bh[cur][col][8 * lh];
      v8bf p1 = *(const v8bf*)&Bh[cur][col][16 + 8 * lh];
      bfh[nt] = __builtin_shufflevector(p0, p1, 0,1,2,3,4,5,6,7,8,9,10,11,12,13,14,15);
      v8bf q0 = *(const v8bf*)&Bl[cur][col][8 * lh];
      v8bf q1 = *(const v8bf*)&Bl[cur][col][16 + 8 * lh];
      bfl[nt] = __builtin_shufflevector(q0, q1, 0,1,2,3,4,5,6,7,8,9,10,11,12,13,14,15);
    }

    // ---- 24 WMMAs: hi*hi + hi*lo + lo*hi (split-bf16 ~ fp32) ----
#pragma unroll
    for (int mt = 0; mt < 2; ++mt)
#pragma unroll
      for (int nt = 0; nt < 4; ++nt) {
        acc[mt][nt] = __builtin_amdgcn_wmma_f32_16x16x32_bf16(
            false, afh[mt], false, bfh[nt], (short)0, acc[mt][nt], false, false);
        acc[mt][nt] = __builtin_amdgcn_wmma_f32_16x16x32_bf16(
            false, afh[mt], false, bfl[nt], (short)0, acc[mt][nt], false, false);
        acc[mt][nt] = __builtin_amdgcn_wmma_f32_16x16x32_bf16(
            false, afl[mt], false, bfh[nt], (short)0, acc[mt][nt], false, false);
      }

    if (k + 1 < nk) sstore(cur ^ 1);       // fill the other buffer
    __syncthreads();                        // single barrier per chunk
  }

  // ---- epilogue: bias (+relu), C/D layout: VGPR r -> row r+8*lh, col lm ----
#pragma unroll
  for (int nt = 0; nt < 4; ++nt) {
    int col = bn0 + wn * 64 + nt * 16 + lm;
    float bv = bias[col];
#pragma unroll
    for (int mt = 0; mt < 2; ++mt) {
      int rbase = bm0 + wm * 32 + mt * 16 + 8 * lh;
#pragma unroll
      for (int r = 0; r < 8; ++r) {
        float v = acc[mt][nt][r] + bv;
        if (do_relu) v = fmaxf(v, 0.f);
        C[(long)(rbase + r) * N + col] = v;
      }
    }
  }
}

// --------------------------------------------------------------- combine ----
__global__ __launch_bounds__(256) void combine_kernel(
    const float* __restrict__ o2, const float* __restrict__ gate,
    const int* __restrict__ eidx, const int* __restrict__ pos,
    float* __restrict__ out)
{
  int t = blockIdx.x;
  int p = pos[t];
  int e = eidx[t];
  float g = gate[t];
  float4 r = {0.f, 0.f, 0.f, 0.f};
  if (p < CAP) {
    const float4* src = (const float4*)(o2 + (long)(e * CAP + p) * DIMD);
    float4 v = src[threadIdx.x];
    r = float4{g * v.x, g * v.y, g * v.z, g * v.w};
  }
  ((float4*)(out + (long)t * DIMD))[threadIdx.x] = r;
}

// ---------------------------------------------------------------------------
extern "C" void kernel_launch(void* const* d_in, const int* in_sizes, int n_in,
                              void* d_out, int out_size, void* d_ws, size_t ws_size,
                              hipStream_t stream) {
  (void)in_sizes; (void)n_in; (void)out_size; (void)ws_size;
  const float* x   = (const float*)d_in[0];  // [B,S,D]
  const float* rw  = (const float*)d_in[1];  // [D,E]
  const float* w1  = (const float*)d_in[2];  // [E,D,F]
  const float* b1  = (const float*)d_in[3];  // [E,F]
  const float* w2  = (const float*)d_in[4];  // [E,F,D]
  const float* b2  = (const float*)d_in[5];  // [E,D]
  float* out = (float*)d_out;                // [B,S,D]

  char* ws = (char*)d_ws;
  float* gate = (float*)ws;  ws += (size_t)T_TOK * 4;
  int*   eidx = (int*)ws;    ws += (size_t)T_TOK * 4;
  int*   pos  = (int*)ws;    ws += (size_t)T_TOK * 4;
  int*   slot = (int*)ws;    ws += (size_t)NEXP * CAP * 4;
  float* xe   = (float*)ws;  ws += (size_t)NEXP * CAP * DIMD * 4;  // 32 MB
  float* o2   = (float*)ws;  ws += (size_t)NEXP * CAP * DIMD * 4;  // 32 MB
  float* hbuf = (float*)ws;  // NEXP*CAP*DIMF*4 = 134 MB

  router_kernel<<<T_TOK / 8, 256, 0, stream>>>(x, rw, gate, eidx);
  scan_kernel<<<1, 256, 0, stream>>>(eidx, pos, slot);
  gather_kernel<<<NEXP * CAP, 256, 0, stream>>>(x, slot, xe);
  // H = relu(Xe * W1 + b1):  M=CAP, N=DIMF, K=DIMD
  gemm_kernel<<<dim3(DIMF / BN, CAP / BM, NEXP), 256, 0, stream>>>(
      xe, w1, hbuf, b1, DIMF, DIMD,
      (long)CAP * DIMD, (long)DIMD * DIMF, (long)CAP * DIMF, DIMF, 1);
  // O2 = H * W2 + b2:  M=CAP, N=DIMD, K=DIMF
  gemm_kernel<<<dim3(DIMD / BN, CAP / BM, NEXP), 256, 0, stream>>>(
      hbuf, w2, o2, b2, DIMD, DIMF,
      (long)CAP * DIMF, (long)DIMF * DIMD, (long)CAP * DIMD, DIMD, 0);
  combine_kernel<<<T_TOK, 256, 0, stream>>>(o2, gate, eidx, pos, out);
}
